// AttentionHead_8486855377263
// MI455X (gfx1250) — matrix operand order repaired and checked by reading, not compiled, not run on hardware
//
#include <hip/hip_runtime.h>

#define B_ 8
#define T_ 2048
#define C_ 1024
#define H_ 64
#define KSTEPS 32       // C_/32

typedef __bf16 bf16;
typedef __attribute__((ext_vector_type(16))) __bf16 v16bf;
typedef __attribute__((ext_vector_type(8)))  float  v8f;

union V16 { v16bf v; uint4 q[2]; bf16 e[16]; };
union V8B { uint4 q;  bf16 e[8]; };

static __device__ __forceinline__ v8f wmma_bf16(v16bf a, v16bf b, v8f c) {
  // D = A(16x32 bf16) * B(32x16 bf16) + C(16x16 f32)
  return __builtin_amdgcn_wmma_f32_16x16x32_bf16(false, a, false, b, (short)0, c, false, false);
}

// ---------------------------------------------------------------------------
// Kernel 1: pack weights fp32 -> bf16 in WMMA B-fragment order.
// wpack flat index = mat*65536 + kk*2048 + ct*512 + lane*16 + j
// element(lane,j) = w[ kk*32 + (lane/16)*8 + (j%8) + (j/8)*16 ][ ct*16 + lane%16 ]
// q-scale (1/32) folded into wq here.
// ---------------------------------------------------------------------------
__global__ __launch_bounds__(256) void pack_w_kernel(
    const float* __restrict__ wq, const float* __restrict__ wk,
    const float* __restrict__ wv, bf16* __restrict__ wpack) {
  int idx  = blockIdx.x * 256 + threadIdx.x;           // 0..196607
  int j    = idx & 15;
  int lane = (idx >> 4) & 31;
  int ct   = (idx >> 9) & 3;
  int kk   = (idx >> 11) & 31;
  int mat  = idx >> 16;                                // 0=q 1=k 2=v
  int row  = kk * 32 + (lane >> 4) * 8 + (j & 7) + (j >> 3) * 16;
  int col  = ct * 16 + (lane & 15);
  const float* w = (mat == 0) ? wq : ((mat == 1) ? wk : wv);
  float vfl = w[row * H_ + col];
  if (mat == 0) vfl *= 0.03125f;                       // 1/sqrt(1024)
  wpack[idx] = (bf16)vfl;
}

// ---------------------------------------------------------------------------
// Kernel 2: QKV projection. One wave per (16-row tile, matrix).
// q,k stored bf16 row-major [16384][64]; v stored transposed [B][64][2048].
// ---------------------------------------------------------------------------
__global__ __launch_bounds__(128) void proj_kernel(
    const float* __restrict__ x, const bf16* __restrict__ wpack,
    const float* __restrict__ bq, const float* __restrict__ bk,
    const float* __restrict__ bv,
    bf16* __restrict__ qws, bf16* __restrict__ kws, bf16* __restrict__ vT) {
  const int lane = threadIdx.x & 31;
  const int n    = lane & 15;
  const int sel  = lane >> 4;
  const int gw   = blockIdx.x * 4 + (threadIdx.x >> 5); // 0..3071
  const int mat  = gw % 3;
  const int tile = gw / 3;                              // 0..1023
  const int t0   = tile * 16;

  const float* xrow = x + (size_t)(t0 + n) * C_;        // A-frag row = lane%16
  const bf16*  wp   = wpack + mat * (KSTEPS * 4 * 32 * 16);

  v8f acc[4] = {{}, {}, {}, {}};
  for (int kk = 0; kk < KSTEPS; ++kk) {
    const int kb = kk * 32 + sel * 8;
    float4 f0 = *(const float4*)(xrow + kb);
    float4 f1 = *(const float4*)(xrow + kb + 4);
    float4 f2 = *(const float4*)(xrow + kb + 16);
    float4 f3 = *(const float4*)(xrow + kb + 20);
    V16 a;
    a.e[0] = (bf16)f0.x; a.e[1] = (bf16)f0.y; a.e[2]  = (bf16)f0.z; a.e[3]  = (bf16)f0.w;
    a.e[4] = (bf16)f1.x; a.e[5] = (bf16)f1.y; a.e[6]  = (bf16)f1.z; a.e[7]  = (bf16)f1.w;
    a.e[8] = (bf16)f2.x; a.e[9] = (bf16)f2.y; a.e[10] = (bf16)f2.z; a.e[11] = (bf16)f2.w;
    a.e[12]= (bf16)f3.x; a.e[13]= (bf16)f3.y; a.e[14] = (bf16)f3.z; a.e[15] = (bf16)f3.w;
#pragma unroll
    for (int ct = 0; ct < 4; ++ct) {
      const bf16* bp = wp + ((kk * 4 + ct) * 32 + lane) * 16;
      V16 bfr;
      bfr.q[0] = *(const uint4*)(bp);
      bfr.q[1] = *(const uint4*)(bp + 8);
      acc[ct] = wmma_bf16(a.v, bfr.v, acc[ct]);
    }
  }

  const float* bias = (mat == 0) ? bq : ((mat == 1) ? bk : bv);
  const float bscale = (mat == 0) ? 0.03125f : 1.0f;
#pragma unroll
  for (int ct = 0; ct < 4; ++ct) {
    const int col = ct * 16 + n;
    const float bval = bias[col] * bscale;
    if (mat == 2) {
      // store transposed: vT[b][col][t], lane's 8 rows are consecutive t
      const int bidx = t0 / T_;
      const int tl   = (t0 % T_) + 8 * sel;
      V8B pk;
#pragma unroll
      for (int r = 0; r < 8; ++r) pk.e[r] = (bf16)(acc[ct][r] + bval);
      *(uint4*)(vT + ((size_t)(bidx * H_ + col) * T_) + tl) = pk.q;
    } else {
      bf16* dst = (mat == 0) ? qws : kws;
#pragma unroll
      for (int r = 0; r < 8; ++r) {
        const int t = t0 + 8 * sel + r;                 // D layout: M = r + 8*sel
        dst[(size_t)t * H_ + col] = (bf16)(acc[ct][r] + bval);
      }
    }
  }
}

// ---------------------------------------------------------------------------
// Kernel 3: flash attention with 4-way key split.
// Wave r of a 4-wave workgroup processes key-pairs p == r (mod 4) for a
// query tile; partial (m, l, O^T) states merged through LDS by wave 0.
// S^T = K * Q^T  (A=K tile, B=Q^T)  ->  softmax per-lane  ->  O^T = V^T * P^T
// ---------------------------------------------------------------------------
static __device__ __forceinline__ void attend_split(
    int b, int qt, int parity,
    const bf16* __restrict__ qws, const bf16* __restrict__ kws,
    const bf16* __restrict__ vT,
    float& m_out, float& l_out, v8f oacc[4]) {
  const int lane = threadIdx.x & 31;
  const int n    = lane & 15;   // = query column of all D tiles
  const int sel  = lane >> 4;
  const int q0   = qt * 16;
  const int qi   = q0 + n;

  // B-fragments of Q^T (element j <-> h = hk*32 + sel*8 + (j%8) + (j/8)*16)
  const bf16* qp = qws + (size_t)(b * T_ + qi) * H_;
  V16 bq0, bq1;
  bq0.q[0] = *(const uint4*)(qp + sel * 8);
  bq0.q[1] = *(const uint4*)(qp + 16 + sel * 8);
  bq1.q[0] = *(const uint4*)(qp + 32 + sel * 8);
  bq1.q[1] = *(const uint4*)(qp + 48 + sel * 8);

  oacc[0] = (v8f){}; oacc[1] = (v8f){}; oacc[2] = (v8f){}; oacc[3] = (v8f){};
  float m_state = -__builtin_inff();
  float l_state = 0.0f;

  const int npairs = (qt + 2) >> 1;                     // 32-key steps
  for (int p = parity; p < npairs; p += 4) {
    const int k0 = p * 32;
    if (p + 4 < npairs)
      __builtin_prefetch(kws + (size_t)(b * T_ + k0 + 128 + n) * H_, 0, 3);

    // A-fragments of K tiles (row m = key = k0(+16) + lane%16)
    const bf16* kpA = kws + (size_t)(b * T_ + k0 + n) * H_;
    const bf16* kpB = kws + (size_t)(b * T_ + k0 + 16 + n) * H_;
    V16 aA0, aA1, aB0, aB1;
    aA0.q[0] = *(const uint4*)(kpA + sel * 8);
    aA0.q[1] = *(const uint4*)(kpA + 16 + sel * 8);
    aA1.q[0] = *(const uint4*)(kpA + 32 + sel * 8);
    aA1.q[1] = *(const uint4*)(kpA + 48 + sel * 8);
    aB0.q[0] = *(const uint4*)(kpB + sel * 8);
    aB0.q[1] = *(const uint4*)(kpB + 16 + sel * 8);
    aB1.q[0] = *(const uint4*)(kpB + 32 + sel * 8);
    aB1.q[1] = *(const uint4*)(kpB + 48 + sel * 8);

    v8f sA = {}, sB = {};
    sA = wmma_bf16(aA0.v, bq0.v, sA);
    sA = wmma_bf16(aA1.v, bq1.v, sA);
    sB = wmma_bf16(aB0.v, bq0.v, sB);
    sB = wmma_bf16(aB1.v, bq1.v, sB);

    if (p == npairs - 1) {                              // causal mask, diag pair
#pragma unroll
      for (int r = 0; r < 8; ++r) {
        const int keyA = k0 + 8 * sel + r;
        sA[r] = (keyA <= qi) ? sA[r] : -__builtin_inff();
        sB[r] = (keyA + 16 <= qi) ? sB[r] : -__builtin_inff();
      }
    }

    float mx = -__builtin_inff();
#pragma unroll
    for (int r = 0; r < 8; ++r) mx = fmaxf(mx, fmaxf(sA[r], sB[r]));
    mx = fmaxf(mx, __shfl_xor(mx, 16, 32));
    const float m_new = fmaxf(m_state, mx);
    const float alpha = __expf(m_state - m_new);

    float pA[8], pB[8], rs = 0.0f;
#pragma unroll
    for (int r = 0; r < 8; ++r) {
      pA[r] = __expf(sA[r] - m_new);
      pB[r] = __expf(sB[r] - m_new);
      rs += pA[r] + pB[r];
    }
    rs += __shfl_xor(rs, 16, 32);
    l_state = l_state * alpha + rs;
    m_state = m_new;
#pragma unroll
    for (int ht = 0; ht < 4; ++ht)
#pragma unroll
      for (int r = 0; r < 8; ++r) oacc[ht][r] *= alpha;

    // Build B-fragment of P^T: element j <-> key = k0 + sel*16 + j.
    // Lane-half exchange: half0 needs keys 8..15 (held by half1 tileA),
    // half1 needs keys 16..23 (held by half0 tileB).
    V16 bP;
#pragma unroll
    for (int r = 0; r < 8; ++r) {
      const float send = sel ? pA[r] : pB[r];
      const float recv = __shfl_xor(send, 16, 32);
      bP.e[r]     = (bf16)(sel ? recv : pA[r]);
      bP.e[8 + r] = (bf16)(sel ? pB[r] : recv);
    }

    // O^T += V^T * P^T  (A-frag of V^T: row m = h_local = lane%16)
#pragma unroll
    for (int ht = 0; ht < 4; ++ht) {
      const bf16* vph = vT + ((size_t)(b * H_ + ht * 16 + n) * T_) + k0;
      V16 aV;
      aV.q[0] = *(const uint4*)(vph + sel * 16);
      aV.q[1] = *(const uint4*)(vph + sel * 16 + 8);
      oacc[ht] = wmma_bf16(aV.v, bP.v, oacc[ht]);
    }
  }
  m_out = m_state;
  l_out = l_state;
}

__global__ __launch_bounds__(128) void attn_kernel(
    const bf16* __restrict__ qws, const bf16* __restrict__ kws,
    const bf16* __restrict__ vT, float* __restrict__ out) {
  __shared__ float sm[3][32];
  __shared__ float sl[3][32];
  __shared__ float so[3][32][32];   // [partial][ht*8+r][lane]

  const int lane   = threadIdx.x & 31;
  const int parity = threadIdx.x >> 5;                  // wave in block: 0..3
  const int n      = lane & 15;
  const int sel    = lane >> 4;
  const int b      = blockIdx.x >> 6;                   // 64 blocks per batch
  const int w      = blockIdx.x & 63;

#pragma unroll
  for (int round = 0; round < 2; ++round) {
    const int qt = round ? (127 - w) : w;               // load-balanced pair
    float m1, l1;
    v8f oacc[4];
    attend_split(b, qt, parity, qws, kws, vT, m1, l1, oacc);

    if (parity != 0) {                                  // spill partial to LDS
      sm[parity - 1][lane] = m1;
      sl[parity - 1][lane] = l1;
#pragma unroll
      for (int ht = 0; ht < 4; ++ht)
#pragma unroll
        for (int r = 0; r < 8; ++r) so[parity - 1][ht * 8 + r][lane] = oacc[ht][r];
    }
    __syncthreads();

    if (parity == 0) {                                  // merge 4 flash partials
      float mm[4], ll[4];
      mm[0] = m1; ll[0] = l1;
#pragma unroll
      for (int i = 0; i < 3; ++i) { mm[i + 1] = sm[i][lane]; ll[i + 1] = sl[i][lane]; }
      float mN = mm[0];
#pragma unroll
      for (int i = 1; i < 4; ++i) mN = fmaxf(mN, mm[i]);
      float wgt[4], lsum = 0.0f;
#pragma unroll
      for (int i = 0; i < 4; ++i) { wgt[i] = __expf(mm[i] - mN); lsum += ll[i] * wgt[i]; }
      const float inv = 1.0f / lsum;

      const int qi = qt * 16 + n;
#pragma unroll
      for (int ht = 0; ht < 4; ++ht) {
        float ov[8];
#pragma unroll
        for (int r = 0; r < 8; ++r) {
          float acc = oacc[ht][r] * wgt[0];
#pragma unroll
          for (int i = 0; i < 3; ++i) acc += so[i][ht * 8 + r][lane] * wgt[i + 1];
          ov[r] = acc * inv;
        }
        float* op = out + (size_t)(b * T_ + qi) * H_ + ht * 16 + sel * 8;
        float4 lo = { ov[0], ov[1], ov[2], ov[3] };
        float4 hi = { ov[4], ov[5], ov[6], ov[7] };
        *(float4*)(op)     = lo;
        *(float4*)(op + 4) = hi;
      }
    }
    __syncthreads();                                    // LDS reuse next round
  }
}

// ---------------------------------------------------------------------------
extern "C" void kernel_launch(void* const* d_in, const int* in_sizes, int n_in,
                              void* d_out, int out_size, void* d_ws, size_t ws_size,
                              hipStream_t stream) {
  const float* x  = (const float*)d_in[0];
  const float* wq = (const float*)d_in[1];
  const float* bq = (const float*)d_in[2];
  const float* wk = (const float*)d_in[3];
  const float* bk = (const float*)d_in[4];
  const float* wv = (const float*)d_in[5];
  const float* bv = (const float*)d_in[6];
  float* out = (float*)d_out;

  bf16* qws   = (bf16*)d_ws;                 // 16384*64 bf16 = 2 MB
  bf16* kws   = qws + (size_t)B_ * T_ * H_;  // 2 MB
  bf16* vT    = kws + (size_t)B_ * T_ * H_;  // 2 MB (transposed [B][H][T])
  bf16* wpack = vT  + (size_t)B_ * T_ * H_;  // 3*65536 bf16 = 384 KB

  pack_w_kernel<<<768, 256, 0, stream>>>(wq, wk, wv, wpack);
  proj_kernel<<<768, 128, 0, stream>>>(x, wpack, bq, bk, bv, qws, kws, vT);
  attn_kernel<<<512, 128, 0, stream>>>(qws, kws, vT, out);  // 2048 waves
}
